// DeepInterestEvolutionNetwork_51831665328188
// MI455X (gfx1250) — matrix operand order repaired and checked by reading, not compiled, not run on hardware
//
#include <hip/hip_runtime.h>

typedef __attribute__((ext_vector_type(16))) _Float16 v16h;
typedef __attribute__((ext_vector_type(8)))  _Float16 v8h;
typedef __attribute__((ext_vector_type(8)))  float    v8f;
typedef __attribute__((ext_vector_type(4)))  unsigned int v4u;
typedef __attribute__((ext_vector_type(4)))  int      v4i;
typedef __attribute__((ext_vector_type(8)))  int      v8i;

#define B_  2048
#define T_  200
#define E_  128
#define H_  128
#define G3  384
#define BM  16
#define THREADS 256

__device__ __forceinline__ v8f wmma_f16(v16h a, v16h b, v8f c) {
  return __builtin_amdgcn_wmma_f32_16x16x32_f16(false, a, false, b, (short)0, c, false, false);
}

// A fragment 16x32 f16 from row-major LDS (row stride ld, K offset k0).
// Per lane the K-run kb..kb+15 is contiguous -> 2x ds_load_b128.
__device__ __forceinline__ v16h load_a(const _Float16* A, int ld, int k0, int lane) {
  const _Float16* p = A + (lane & 15) * ld + k0 + ((lane >> 4) << 4);
  union { v16h v; v8h h[2]; } u;
  u.h[0] = *(const v8h*)(p);
  u.h[1] = *(const v8h*)(p + 8);
  return u.v;
}

// B fragment from fragment-swizzled weight image: lane's 16 halves contiguous.
__device__ __forceinline__ v16h load_b_sw(const _Float16* sWsw, int f, int lane) {
  const v8h* q = (const v8h*)(sWsw + ((size_t)(f * 32 + lane)) * 16);
  union { v16h v; v8h h[2]; } u;
  u.h[0] = q[0];
  u.h[1] = q[1];
  return u.v;
}

// B fragment 32x16 from row-major f32 global [K][N] (setup-only)
__device__ __forceinline__ v16h load_b_f32(const float* Bm, int ld, int k0, int n0, int lane) {
  const float* p = Bm + (size_t)(k0 + ((lane >> 4) << 4)) * ld + n0 + (lane & 15);
  v16h b;
#pragma unroll
  for (int v = 0; v < 8; ++v) {
    b[2 * v]     = (_Float16)p[(size_t)(2 * v) * ld];
    b[2 * v + 1] = (_Float16)p[(size_t)(2 * v + 1) * ld];
  }
  return b;
}

// B fragment = W^T, W row-major f32 global [N][K] (setup-only)
__device__ __forceinline__ v16h load_bt_f32(const float* W, int ld, int k0, int n0, int lane) {
  const float* p = W + (size_t)(n0 + (lane & 15)) * ld + k0 + ((lane >> 4) << 4);
  v16h b;
#pragma unroll
  for (int v = 0; v < 8; ++v) {
    b[2 * v]     = (_Float16)p[2 * v];
    b[2 * v + 1] = (_Float16)p[2 * v + 1];
  }
  return b;
}

// C/D f32 16x16: VGPR v -> row v + 8*(lane>>4), col n0 + (lane&15)
__device__ __forceinline__ void store_c(float* D, int ld, int n0, v8f c, int lane) {
  float* p = D + ((lane >> 4) << 3) * ld + n0 + (lane & 15);
#pragma unroll
  for (int v = 0; v < 8; ++v) p[v * ld] = c[v];
}

// ---------- Tensor Data Mover: 2D tile load, optional LDS row padding ----------
__device__ __forceinline__ void tdm_load_2d(const void* gaddr, unsigned lds_off,
                                            unsigned data_size_code,   // 0=1B,1=2B,2=4B
                                            unsigned tile_d0, unsigned tile_d1,
                                            unsigned tensor_d0, unsigned tensor_d1,
                                            unsigned long long stride0,
                                            unsigned pad_en, unsigned pad_interval,
                                            unsigned pad_amount) {
  unsigned long long ga = (unsigned long long)gaddr;
  v4u g0;
  g0[0] = 1u;                                              // count=1 (valid user D#)
  g0[1] = lds_off;                                         // lds_addr (bytes)
  g0[2] = (unsigned)(ga & 0xFFFFFFFFu);                    // global_addr[31:0]
  g0[3] = (unsigned)((ga >> 32) & 0x1FFFFFFu) | (2u << 30);// global_addr[56:32], type=2
  v8i g1;
  g1[0] = (int)((data_size_code << 16) | (pad_en << 20) |
                (pad_interval << 22) | (pad_amount << 25));
  g1[1] = (int)((tensor_d0 & 0xFFFFu) << 16);              // tensor_dim0[15:0]
  g1[2] = (int)(((tensor_d0 >> 16) & 0xFFFFu) | ((tensor_d1 & 0xFFFFu) << 16));
  g1[3] = (int)(((tensor_d1 >> 16) & 0xFFFFu) | ((tile_d0 & 0xFFFFu) << 16));
  g1[4] = (int)(tile_d1 & 0xFFFFu);                        // tile_dim1 (tile_dim2=0)
  g1[5] = (int)(unsigned)(stride0 & 0xFFFFFFFFull);        // tensor_dim0_stride[31:0]
  g1[6] = (int)(unsigned)((stride0 >> 32) & 0xFFFFull);    // [47:32]
  g1[7] = 0;
  v4i z4 = {0, 0, 0, 0};
#if __clang_major__ >= 23
  v8i z8 = {0, 0, 0, 0, 0, 0, 0, 0};
  __builtin_amdgcn_tensor_load_to_lds(g0, g1, z4, z4, z8, 0);
#else
  __builtin_amdgcn_tensor_load_to_lds(g0, g1, z4, z4, 0);
#endif
}

// ---- LDS carve-out (dynamic, 287808 bytes) ----
#define OFF_W      0          // 192 frags * 32 lanes * 16 halves (swizzled weights, both phases)
#define OFF_XH     196608     // [16][256] f16: cols 0-127 = x, 128-255 = h (or q / r*h)
#define OFF_G      204800     // [16][384] f32 gate scratch (AUGRU: zr [16][256] + n [16][128])
#define OFF_GHN    229376     // [16][128] f32 (gh n-gate)
#define OFF_H      237568     // [16][128] f32 hidden state
#define OFF_QK     245760     // [16][128] f32 qk = q @ Wk^T
#define OFF_SCORE  253952     // [200][16] f32 attention scores
#define OFF_BIH    266752
#define OFF_BHH    268288
#define OFF_BZ     269824
#define OFF_BR     270336
#define OFF_BN     270848
#define OFF_SB     271360
#define OFF_XF32   271424     // double-buffered TDM x landing pad: 2 * [16][128] f32
#define SMEM_BYTES 287808

__global__ void __launch_bounds__(THREADS) dien_fused_kernel(
    const float* __restrict__ target, const float* __restrict__ hist,
    const float* __restrict__ Wih, const float* __restrict__ bih_g,
    const float* __restrict__ Whh, const float* __restrict__ bhh_g,
    const float* __restrict__ Wq, const float* __restrict__ bq,
    const float* __restrict__ Wk, const float* __restrict__ bk,
    const float* __restrict__ Wz, const float* __restrict__ bz_g,
    const float* __restrict__ Wr, const float* __restrict__ br_g,
    const float* __restrict__ Wn, const float* __restrict__ bn_g,
    float* __restrict__ out, _Float16* __restrict__ gout) {
  extern __shared__ char smem[];
  _Float16* sW    = (_Float16*)(smem + OFF_W);
  _Float16* sXH   = (_Float16*)(smem + OFF_XH);
  float*    sG    = (float*)(smem + OFF_G);
  float*    sGhn  = (float*)(smem + OFF_GHN);
  float*    sH    = (float*)(smem + OFF_H);
  float*    sQK   = (float*)(smem + OFF_QK);
  float*    sScore= (float*)(smem + OFF_SCORE);
  float*    sBih  = (float*)(smem + OFF_BIH);
  float*    sBhh  = (float*)(smem + OFF_BHH);
  float*    sBz   = (float*)(smem + OFF_BZ);
  float*    sBr   = (float*)(smem + OFF_BR);
  float*    sBn   = (float*)(smem + OFF_BN);
  float*    sSB   = (float*)(smem + OFF_SB);

  const int tid  = threadIdx.x;
  const int lane = tid & 31;
  const int w    = tid >> 5;
  const int b0   = blockIdx.x * BM;
  const int m    = tid >> 4;       // batch row within tile
  const int c0   = (tid & 15) * 8; // 8 contiguous cols per thread

  // ===== setup: swizzle GRU weights into fragment order, biases, target tile =====
  // GRU frag id: f = (mat*24 + nt)*4 + kk ; lane strip = col (nt*16+l&15), rows kb..kb+15
  for (int idx = tid; idx < 192 * 32; idx += THREADS) {
    int f = idx >> 5, l = idx & 31;
    int mat = f / 96, rem = f - mat * 96;
    int nt = rem >> 2, kk = rem & 3;
    int col = nt * 16 + (l & 15);
    int kb  = kk * 32 + ((l >> 4) << 4);
    const float* Wsrc = mat ? Whh : Wih;
    _Float16* dst = sW + (size_t)idx * 16;
#pragma unroll
    for (int e = 0; e < 16; ++e)
      dst[e] = (_Float16)Wsrc[(size_t)(kb + e) * G3 + col];
  }
  for (int i = tid; i < G3; i += THREADS) { sBih[i] = bih_g[i]; sBhh[i] = bhh_g[i]; }
  {
    const float4* src = (const float4*)(target + (size_t)(b0 + m) * E_ + c0);
    float4 x0 = src[0], x1 = src[1];
    union { _Float16 h[8]; uint4 u; } xx;
    xx.h[0] = (_Float16)x0.x; xx.h[1] = (_Float16)x0.y; xx.h[2] = (_Float16)x0.z; xx.h[3] = (_Float16)x0.w;
    xx.h[4] = (_Float16)x1.x; xx.h[5] = (_Float16)x1.y; xx.h[6] = (_Float16)x1.z; xx.h[7] = (_Float16)x1.w;
    *(uint4*)(sXH + m * 256 + c0) = xx.u;
  }
  __syncthreads();

  // q = target @ Wq (1 N-tile per wave) -> sG scratch (stride 128)
  {
    int n0 = w * 16;
    v8f acc = {};
#pragma unroll
    for (int kk = 0; kk < 4; ++kk) {
      v16h a = load_a(sXH, 256, kk * 32, lane);
      v16h b = load_b_f32(Wq, H_, kk * 32, n0, lane);
      acc = wmma_f16(a, b, acc);
    }
    store_c(sG, 128, n0, acc, lane);
  }
  __syncthreads();
  // add bq; stash q f16 at sXH[:,128:]; sb[m] = q[m].bk
  {
    float qv[8];
#pragma unroll
    for (int i = 0; i < 8; ++i) {
      qv[i] = sG[m * 128 + c0 + i] + bq[c0 + i];
      sXH[m * 256 + 128 + c0 + i] = (_Float16)qv[i];
    }
    float p = 0.f;
#pragma unroll
    for (int i = 0; i < 8; ++i) p += qv[i] * bk[c0 + i];
    p += __shfl_xor(p, 1, 16); p += __shfl_xor(p, 2, 16);
    p += __shfl_xor(p, 4, 16); p += __shfl_xor(p, 8, 16);
    if ((tid & 15) == 0) sSB[m] = p;
  }
  __syncthreads();
  // qk = q @ Wk^T
  {
    int n0 = w * 16;
    v8f acc = {};
#pragma unroll
    for (int kk = 0; kk < 4; ++kk) {
      v16h a = load_a(sXH + 128, 256, kk * 32, lane);
      v16h b = load_bt_f32(Wk, H_, kk * 32, n0, lane);
      acc = wmma_f16(a, b, acc);
    }
    store_c(sQK, 128, n0, acc, lane);
  }
  __syncthreads();
#pragma unroll
  for (int i = 0; i < 8; ++i) {
    sH[m * H_ + c0 + i] = 0.f;
    sXH[m * 256 + 128 + c0 + i] = (_Float16)0.f;
  }
  __syncthreads();

  // ===== GRU over T (TDM-fed, double buffered), attention fused =====
  const float inv_sqrt_e = 0.088388347648318447f; // 1/sqrt(128)
  if (w == 0)
    tdm_load_2d(hist + (size_t)b0 * T_ * E_, OFF_XF32, /*f32*/2,
                E_, BM, E_, 1u << 20, (unsigned long long)T_ * E_, 0, 0, 0);
  for (int t = 0; t < T_; ++t) {
    if (w == 0) __builtin_amdgcn_s_wait_tensorcnt((short)0);
    __syncthreads();
    // convert TDM f32 tile -> f16 x columns of sXH
    {
      const float* xb = (const float*)(smem + OFF_XF32 + (t & 1) * (BM * E_ * 4));
      const float4* src = (const float4*)(xb + m * E_ + c0);
      float4 x0 = src[0], x1 = src[1];
      union { _Float16 h[8]; uint4 u; } xx;
      xx.h[0] = (_Float16)x0.x; xx.h[1] = (_Float16)x0.y; xx.h[2] = (_Float16)x0.z; xx.h[3] = (_Float16)x0.w;
      xx.h[4] = (_Float16)x1.x; xx.h[5] = (_Float16)x1.y; xx.h[6] = (_Float16)x1.z; xx.h[7] = (_Float16)x1.w;
      *(uint4*)(sXH + m * 256 + c0) = xx.u;
    }
    if (w == 0 && t + 1 < T_)   // overlap next DMA with this step's GEMMs
      tdm_load_2d(hist + (size_t)b0 * T_ * E_ + (size_t)(t + 1) * E_,
                  OFF_XF32 + ((t + 1) & 1) * (BM * E_ * 4), 2,
                  E_, BM, E_, 1u << 20, (unsigned long long)T_ * E_, 0, 0, 0);
    __syncthreads();
    // gi = x @ Wih, gh = h @ Whh : A frags hoisted, 3 N-tiles per wave
    {
      v16h ax[4], ah[4];
#pragma unroll
      for (int kk = 0; kk < 4; ++kk) {
        ax[kk] = load_a(sXH,       256, kk * 32, lane);
        ah[kk] = load_a(sXH + 128, 256, kk * 32, lane);
      }
#pragma unroll
      for (int j = 0; j < 3; ++j) {
        int nt = w * 3 + j, n0 = nt * 16;
        v8f gi = {}, gh = {};
#pragma unroll
        for (int kk = 0; kk < 4; ++kk) {
          gi = wmma_f16(ax[kk], load_b_sw(sW, nt * 4 + kk, lane), gi);
          gh = wmma_f16(ah[kk], load_b_sw(sW, 96 + nt * 4 + kk, lane), gh);
        }
        if (n0 < 256) {
          v8f s;
#pragma unroll
          for (int v = 0; v < 8; ++v) s[v] = gi[v] + gh[v];
          store_c(sG, G3, n0, s, lane);
        } else {
          store_c(sG, G3, n0, gi, lane);
          store_c(sGhn, H_, n0 - 256, gh, lane);
        }
      }
    }
    __syncthreads();
    // gates + h update + fused attention dot + gru_out spill
    {
      float part = 0.f;
      union { _Float16 h[8]; uint4 u; } hh;
#pragma unroll
      for (int i = 0; i < 8; ++i) {
        int c = c0 + i;
        float rp = sG[m * G3 + c]       + sBih[c]       + sBhh[c];
        float zp = sG[m * G3 + 128 + c] + sBih[128 + c] + sBhh[128 + c];
        float np = sG[m * G3 + 256 + c] + sBih[256 + c];
        float hn = sGhn[m * H_ + c]     + sBhh[256 + c];
        float r = 1.f / (1.f + __expf(-rp));
        float z = 1.f / (1.f + __expf(-zp));
        float n = tanhf(np + r * hn);
        float h = (1.f - z) * n + z * sH[m * H_ + c];
        sH[m * H_ + c] = h;
        sXH[m * 256 + 128 + c] = (_Float16)h;
        hh.h[i] = (_Float16)h;
        part += h * sQK[m * H_ + c];
      }
      *(uint4*)(gout + ((size_t)t * B_ + b0 + m) * H_ + c0) = hh.u;
      part += __shfl_xor(part, 1, 16); part += __shfl_xor(part, 2, 16);
      part += __shfl_xor(part, 4, 16); part += __shfl_xor(part, 8, 16);
      if ((tid & 15) == 0) sScore[t * BM + m] = (part + sSB[m]) * inv_sqrt_e;
    }
    __syncthreads();
  }

  // ===== re-swizzle AUGRU weights [Wz|Wr|Wn] into fragment order =====
  // frag f = nt*8 + kk (nt 0..23 over 384 cols, kk 0..7 over K=256)
  for (int idx = tid; idx < 192 * 32; idx += THREADS) {
    int f = idx >> 5, l = idx & 31;
    int nt = f >> 3, kk = f & 7;
    int col = nt * 16 + (l & 15);
    int kb  = kk * 32 + ((l >> 4) << 4);
    const float* Wsrc; int cc;
    if (col < 128)      { Wsrc = Wz; cc = col; }
    else if (col < 256) { Wsrc = Wr; cc = col - 128; }
    else                { Wsrc = Wn; cc = col - 256; }
    _Float16* dst = sW + (size_t)idx * 16;
#pragma unroll
    for (int e = 0; e < 16; ++e)
      dst[e] = (_Float16)Wsrc[(size_t)(kb + e) * H_ + cc];
  }
  for (int i = tid; i < H_; i += THREADS) { sBz[i] = bz_g[i]; sBr[i] = br_g[i]; sBn[i] = bn_g[i]; }
  // h0 = gru_out[0] (same thread wrote these values: same-thread RAW)
#pragma unroll
  for (int i = 0; i < 8; ++i) {
    float h = (float)gout[(size_t)(b0 + m) * H_ + c0 + i];
    sH[m * H_ + c0 + i] = h;
    sXH[m * 256 + 128 + c0 + i] = (_Float16)h;
  }
  __syncthreads();

  float* sZR = sG;             // [16][256] z,r pre-activations
  float* sNN = sG + 16 * 256;  // [16][128] n pre-activation

  // ===== AUGRU over T: TDM loads x rows straight into the [x|h] image via LDS padding =====
  for (int t = 0; t < T_; ++t) {
    if (w == 0) {
      // 16 rows of 128 f16 (256B = 64 DWORDs), pad 64 DWORDs after each row
      // -> lands at sXH row stride 512B with h columns untouched
      tdm_load_2d(gout + ((size_t)t * B_ + b0) * H_, OFF_XH, /*f16*/1,
                  H_, BM, H_, 1u << 20, H_, 1, /*64 dwords*/5, /*64 dwords*/63);
      __builtin_amdgcn_s_wait_tensorcnt((short)0);
    }
    __syncthreads();
    // [x|h] @ [Wz|Wr] : K=256, 2 N-tiles per wave, A frags hoisted
    {
      v16h a[8];
#pragma unroll
      for (int kk = 0; kk < 8; ++kk) a[kk] = load_a(sXH, 256, kk * 32, lane);
#pragma unroll
      for (int j = 0; j < 2; ++j) {
        int nt = w * 2 + j;
        v8f acc = {};
#pragma unroll
        for (int kk = 0; kk < 8; ++kk)
          acc = wmma_f16(a[kk], load_b_sw(sW, nt * 8 + kk, lane), acc);
        store_c(sZR, 256, nt * 16, acc, lane);
      }
    }
    __syncthreads();
    // z = sigmoid, r = sigmoid; rebuild A as [x | r*h]
#pragma unroll
    for (int i = 0; i < 8; ++i) {
      int c = c0 + i;
      float z = 1.f / (1.f + __expf(-(sZR[m * 256 + c] + sBz[c])));
      float r = 1.f / (1.f + __expf(-(sZR[m * 256 + 128 + c] + sBr[c])));
      sZR[m * 256 + c] = z;
      sXH[m * 256 + 128 + c] = (_Float16)(r * sH[m * H_ + c]);
    }
    __syncthreads();
    // [x|r*h] @ Wn : K=256, 1 N-tile per wave
    {
      int nt = 16 + w;
      v8f acc = {};
#pragma unroll
      for (int kk = 0; kk < 8; ++kk)
        acc = wmma_f16(load_a(sXH, 256, kk * 32, lane),
                       load_b_sw(sW, nt * 8 + kk, lane), acc);
      store_c(sNN, H_, w * 16, acc, lane);
    }
    __syncthreads();
    // h = (1 - z*a)*h + z*a*tanh(n)
    {
      float a_t = sScore[t * BM + m];
#pragma unroll
      for (int i = 0; i < 8; ++i) {
        int c = c0 + i;
        float n = tanhf(sNN[m * H_ + c] + sBn[c]);
        float z = sZR[m * 256 + c] * a_t;
        float h = (1.f - z) * sH[m * H_ + c] + z * n;
        sH[m * H_ + c] = h;
        sXH[m * 256 + 128 + c] = (_Float16)h;
      }
    }
    __syncthreads();
  }

  // final hidden state -> out [B,H] f32
#pragma unroll
  for (int i = 0; i < 8; ++i)
    out[(size_t)(b0 + m) * H_ + c0 + i] = sH[m * H_ + c0 + i];
}

extern "C" void kernel_launch(void* const* d_in, const int* in_sizes, int n_in,
                              void* d_out, int out_size, void* d_ws, size_t ws_size,
                              hipStream_t stream) {
  (void)in_sizes; (void)n_in; (void)out_size; (void)ws_size;
  const float* target = (const float*)d_in[0];
  const float* hist   = (const float*)d_in[1];
  const float* Wih    = (const float*)d_in[2];
  const float* bih    = (const float*)d_in[3];
  const float* Whh    = (const float*)d_in[4];
  const float* bhh    = (const float*)d_in[5];
  const float* Wq     = (const float*)d_in[6];
  const float* bq     = (const float*)d_in[7];
  const float* Wk     = (const float*)d_in[8];
  const float* bk     = (const float*)d_in[9];
  // d_in[10]/d_in[11] = Wv/bv: unused by the reference forward
  const float* Wz     = (const float*)d_in[12];
  const float* bz     = (const float*)d_in[13];
  const float* Wr     = (const float*)d_in[14];
  const float* br     = (const float*)d_in[15];
  const float* Wn     = (const float*)d_in[16];
  const float* bn     = (const float*)d_in[17];
  float* out = (float*)d_out;
  _Float16* gout = (_Float16*)d_ws;  // gru_out spill: [T][B][H] f16 = 104.9 MB

  hipLaunchKernelGGL(dien_fused_kernel, dim3(B_ / BM), dim3(THREADS), SMEM_BYTES, stream,
                     target, hist, Wih, bih, Whh, bhh, Wq, bq, Wk, bk,
                     Wz, bz, Wr, br, Wn, bn, out, gout);
}
